// SimpleRNN_23811298689963
// MI455X (gfx1250) — matrix operand order repaired and checked by reading, not compiled, not run on hardware
//
#include <hip/hip_runtime.h>

typedef __bf16 bf16;
typedef __attribute__((ext_vector_type(16))) __bf16 v16bf;
typedef __attribute__((ext_vector_type(8)))  __bf16 v8bf;
typedef __attribute__((ext_vector_type(4)))  __bf16 v4bf;
typedef __attribute__((ext_vector_type(8)))  float  v8f;
typedef __attribute__((ext_vector_type(4)))  unsigned int v4u;
typedef __attribute__((ext_vector_type(8)))  int v8i;
typedef __attribute__((ext_vector_type(4)))  int v4i;

#define RNN_B 64
#define RNN_T 512
#define RNN_I 256
#define RNN_H 1024
#define RNN_O 256
#define RNN_NWG 64            // persistent workgroups (one 16-col slice each)

// Publish TDM-written LDS to the optimizer.  The LDS array's address must
// ESCAPE into the asm (plain "memory" clobbers only cover escaped memory, and
// a never-stored, never-escaping LDS global gets its loads folded to poison).
// With the pointer as an input operand + memory clobber, the asm may-write
// through it, so subsequent ds_loads must really be emitted.
#define TDM_PUBLISH(ptr) asm volatile("" :: "v"((const void*)(ptr)) : "memory")

// ---------------------------------------------------------------------------
// WMMA helpers (CDNA5 gfx1250, wave32): D = A(16x32 bf16) * B(32x16 bf16) + C(f32)
// ---------------------------------------------------------------------------
__device__ __forceinline__ v8f wmma_bf16(v16bf a, v16bf b, v8f c) {
  return __builtin_amdgcn_wmma_f32_16x16x32_bf16(
      /*neg_a=*/false, a, /*neg_b=*/false, b,
      /*c_mod=*/(short)0, c, /*reuse_a=*/false, /*reuse_b=*/false);
}

__device__ __forceinline__ v16bf cat8(v8bf lo, v8bf hi) {
  v16bf r;
#pragma unroll
  for (int i = 0; i < 8; ++i) { r[i] = lo[i]; r[i + 8] = hi[i]; }
  return r;
}

// A-matrix 16x32 bf16, row-major source (contiguous k order: k = 16*half + s).
__device__ __forceinline__ v16bf load_A16x32(const bf16* __restrict__ base, int ld,
                                             int m_base, int k, int lane) {
  int row = m_base + (lane & 15);
  int hsel = lane >> 4;
  const bf16* p = base + (size_t)row * ld + k + hsel * 16;
  return cat8(*(const v8bf*)p, *(const v8bf*)(p + 8));
}

// B-matrix 32x16 from row-major W (B[k][n] = W[n][k]); global-memory variant.
__device__ __forceinline__ v16bf load_B32x16(const bf16* __restrict__ Wrows, int ld,
                                             int n_base, int k, int lane) {
  int row = n_base + (lane & 15);
  int kh = k + (lane >> 4) * 16;
  const bf16* p = Wrows + (size_t)row * ld + kh;
  return cat8(*(const v8bf*)p, *(const v8bf*)(p + 8));
}

// ---------------------------------------------------------------------------
// Tensor Data Mover: DMA a 2D bf16 tile (rows x row_elems) from global into
// LDS at byte offset lds_off, with hardware LDS padding of 16B per interval
// (interval = 512B for K<512 rows, 1024B otherwise) to avoid bank conflicts
// on the 16-lane column reads.  D# layout per ISA ch.8 (group0 + group1;
// groups 2/3 zero => dims 2..4 unused).
// ---------------------------------------------------------------------------
__device__ __forceinline__ void tdm_load_2d_bf16(unsigned lds_off,
                                                 const bf16* __restrict__ gsrc,
                                                 unsigned rows, unsigned row_elems,
                                                 unsigned row_stride_elems,
                                                 unsigned pad_interval_code) {
  unsigned long long ga = (unsigned long long)(uintptr_t)gsrc;
  v4u g0 = {0u, 0u, 0u, 0u};
  g0[0] = 1u;                                        // count=1, user D#
  g0[1] = lds_off;                                   // lds_addr (bytes)
  g0[2] = (unsigned)ga;                              // global_addr[31:0]
  g0[3] = (unsigned)((ga >> 32) & 0x01FFFFFFull)     // global_addr[56:32]
        | (2u << 30);                                // type=2 ("image")
  v8i g1 = {0, 0, 0, 0, 0, 0, 0, 0};
  g1[0] = (int)((1u << 16)                           // data_size = 2 bytes
              | (1u << 20)                           // pad_enable
              | (pad_interval_code << 22)            // 6 => 512B, 7 => 1024B
              | (3u << 25));                         // pad_amount code 3 = 4 dw = 16B
  g1[1] = (int)((row_elems & 0xFFFFu) << 16);        // tensor_dim0[15:0]
  g1[2] = (int)((row_elems >> 16) | ((rows & 0xFFFFu) << 16));  // dim0 hi | tensor_dim1 lo
  g1[3] = (int)((rows >> 16) | ((row_elems & 0xFFFFu) << 16));  // dim1 hi | tile_dim0
  g1[4] = (int)(rows & 0xFFFFu);                     // tile_dim1 (tile_dim2 = 0)
  g1[5] = (int)row_stride_elems;                     // tensor_dim0_stride lo32
  v4i z4 = {0, 0, 0, 0};
#if defined(__clang_major__) && __clang_major__ >= 23
  v8i z8 = {0, 0, 0, 0, 0, 0, 0, 0};
  __builtin_amdgcn_tensor_load_to_lds(g0, g1, z4, z4, z8, 0);
#else
  __builtin_amdgcn_tensor_load_to_lds(g0, g1, z4, z4, 0);
#endif
}

// padded LDS row stride (elements) matching the TDM pad settings above
__device__ __host__ constexpr int padded_stride(int K) {
  return K + 8 * (K >= 512 ? 2 : 1);   // +16B per pad interval
}

// ---------------------------------------------------------------------------
// fp32 -> bf16 elementwise convert
// ---------------------------------------------------------------------------
__global__ void f32_to_bf16_kernel(const float* __restrict__ src,
                                   bf16* __restrict__ dst, int n) {
  int i = (blockIdx.x * blockDim.x + threadIdx.x) * 4;
  if (i + 3 < n) {
    float4 v = *(const float4*)(src + i);
    v4bf o;
    o[0] = (bf16)v.x; o[1] = (bf16)v.y; o[2] = (bf16)v.z; o[3] = (bf16)v.w;
    *(v4bf*)(dst + i) = o;
  } else {
    for (; i < n; ++i) dst[i] = (bf16)src[i];
  }
}

// ---------------------------------------------------------------------------
// Zero h0 ping buffer + grid-barrier counter
// ---------------------------------------------------------------------------
__global__ void init_kernel(bf16* __restrict__ h0, unsigned* __restrict__ counter, int n) {
  int i = blockIdx.x * blockDim.x + threadIdx.x;
  if (i < n) h0[i] = (bf16)0.0f;
  if (i == 0) *counter = 0u;
}

// ---------------------------------------------------------------------------
// GEMM with bias: out[M,N](f32) = A[M,K](bf16) @ W[N,K]^T(bf16) + b1[N] + b2[N]
// block = 256 (8 waves); wave tile 32x32 (2x2 WMMA); WG tile 128x64.
// Per-WG 64xK weight slab staged in LDS by the Tensor Data Mover.
// ---------------------------------------------------------------------------
template <int K>
__global__ void __launch_bounds__(256) gemm_bias_kernel(
    const bf16* __restrict__ A, const bf16* __restrict__ W,
    const float* __restrict__ bias1, const float* __restrict__ bias2,
    float* __restrict__ out, int M, int N) {
  constexpr int SSTR = padded_stride(K);
  __shared__ bf16 sB[64 * SSTR];

  const int wave = threadIdx.x >> 5;
  const int lane = threadIdx.x & 31;
  const int m_base = blockIdx.y * 128 + (wave & 3) * 32;
  const int nw_base = (wave >> 2) * 32;                // WG-local column base
  const int n_base = blockIdx.x * 64 + nw_base;        // global column base
  const int colL = lane & 15;
  const int hsel = lane >> 4;

  // TDM: stage W rows [blockIdx.x*64, +64) x K into LDS (sB sits at LDS offset 0)
  if (wave == 0) {
    tdm_load_2d_bf16(0u, W + (size_t)blockIdx.x * 64 * K,
                     /*rows=*/64, /*row_elems=*/K, /*stride=*/K,
                     /*pad_interval_code=*/(K >= 512) ? 7u : 6u);
    __builtin_amdgcn_s_wait_tensorcnt(0);
  }
  __syncthreads();
  TDM_PUBLISH(sB);   // escape sB: its contents are live below this point

  auto load_B_lds = [&](int n_local, int k) -> v16bf {
    int kk = k + hsel * 16;
    int kofs = kk + ((K >= 512 && kk >= 512) ? 8 : 0);   // hop the 16B pad
    const bf16* p = &sB[(n_local + colL) * SSTR + kofs];
    return cat8(*(const v8bf*)p, *(const v8bf*)(p + 8));
  };

  v8f c00 = {}, c01 = {}, c10 = {}, c11 = {};
  const bf16* aRow = A + (size_t)(m_base + colL) * K;
#pragma unroll 8
  for (int k = 0; k < K; k += 32) {
    __builtin_prefetch(aRow + k + 64, 0, 3);             // global_prefetch_b8
    v16bf a0 = load_A16x32(A, K, m_base,      k, lane);
    v16bf a1 = load_A16x32(A, K, m_base + 16, k, lane);
    v16bf b0 = load_B_lds(nw_base,      k);
    v16bf b1 = load_B_lds(nw_base + 16, k);
    c00 = wmma_bf16(a0, b0, c00);
    c01 = wmma_bf16(a0, b1, c01);
    c10 = wmma_bf16(a1, b0, c10);
    c11 = wmma_bf16(a1, b1, c11);
  }

  const int r0l = hsel * 8;
  auto store_tile = [&](v8f c, int mb, int nb) {
    int col = nb + colL;
    float bv = bias1[col] + bias2[col];
#pragma unroll
    for (int v = 0; v < 8; ++v)
      out[(size_t)(mb + r0l + v) * N + col] = c[v] + bv;
  };
  store_tile(c00, m_base,      n_base);
  store_tile(c01, m_base,      n_base + 16);
  store_tile(c10, m_base + 16, n_base);
  store_tile(c11, m_base + 16, n_base + 16);
}

// ---------------------------------------------------------------------------
// Grid-wide barrier (persistent kernel), device-scope atomics, monotonic target
// ---------------------------------------------------------------------------
__device__ __forceinline__ void grid_sync(unsigned* counter, unsigned target) {
  __threadfence();
  __syncthreads();
  if (threadIdx.x == 0) {
    __hip_atomic_fetch_add(counter, 1u, __ATOMIC_ACQ_REL, __HIP_MEMORY_SCOPE_AGENT);
    while (__hip_atomic_load(counter, __ATOMIC_ACQUIRE, __HIP_MEMORY_SCOPE_AGENT) < target) {
      __builtin_amdgcn_s_sleep(2);
    }
  }
  __syncthreads();
}

// ---------------------------------------------------------------------------
// Persistent RNN recurrence: h_t = relu(xp_t + h_{t-1} @ Whh^T)
// 64 WGs x 128 threads. WG owns 16 output columns; Whh slice TDM-loaded into
// LDS once (read from HBM exactly once for all 512 steps). h ping-pongs
// through global bf16 buffers; hseq receives the full (B,T,H) sequence.
// ---------------------------------------------------------------------------
__global__ void __launch_bounds__(128) rnn_persistent_kernel(
    const float* __restrict__ xp,    // (B,T,H) f32: x@Wih^T + biases
    const bf16* __restrict__ Whh,    // (H,H) bf16 row-major
    bf16* __restrict__ hp0,          // ping (B,H), pre-zeroed (h_0 = 0)
    bf16* __restrict__ hp1,          // pong (B,H)
    bf16* __restrict__ hseq,         // (B,T,H) bf16 output sequence
    unsigned* __restrict__ counter, int T) {
  constexpr int SSTR = padded_stride(RNN_H);           // 1040
  __shared__ bf16 sW[16 * SSTR];

  const int n_base = blockIdx.x * 16;
  const int lane = threadIdx.x & 31;
  const int wave = threadIdx.x >> 5;
  const int m_base = wave * 16;        // batch tile (4 waves cover B=64)
  const int colL = lane & 15;
  const int hsel = lane >> 4;
  const unsigned nwg = gridDim.x;

  // TDM: this WG's 16 Whh rows -> LDS (sW at LDS offset 0), padded layout
  if (wave == 0) {
    tdm_load_2d_bf16(0u, Whh + (size_t)n_base * RNN_H,
                     /*rows=*/16, /*row_elems=*/RNN_H, /*stride=*/RNN_H,
                     /*pad_interval_code=*/7u);
    __builtin_amdgcn_s_wait_tensorcnt(0);
  }
  __syncthreads();
  TDM_PUBLISH(sW);   // escape sW: its contents are live below this point

  for (int t = 0; t < T; ++t) {
    const bf16* hp = (t & 1) ? hp1 : hp0;
    bf16* hn = (t & 1) ? hp0 : hp1;

    v8f c = {};
#pragma unroll
    for (int k = 0; k < RNN_H; k += 32) {
      v16bf a = load_A16x32(hp, RNN_H, m_base, k, lane);
      int kk = k + hsel * 16;
      int kofs = kk + ((kk >= 512) ? 8 : 0);           // hop the 16B pad
      const bf16* bp = &sW[colL * SSTR + kofs];
      v16bf b = cat8(*(const v8bf*)bp, *(const v8bf*)(bp + 8));
      c = wmma_bf16(a, b, c);
    }

    const int col = n_base + colL;
    const int r0 = m_base + hsel * 8;
#pragma unroll
    for (int v = 0; v < 8; ++v) {
      int brow = r0 + v;
      float val = c[v] + xp[((size_t)brow * T + t) * RNN_H + col];
      val = fmaxf(val, 0.0f);
      bf16 hv = (bf16)val;
      hn[brow * RNN_H + col] = hv;
      hseq[((size_t)brow * T + t) * RNN_H + col] = hv;
    }
    grid_sync(counter, nwg * (unsigned)(t + 1));
  }
}

// ---------------------------------------------------------------------------
// FC: out[B,O] = h[B,H](bf16) @ fcW[O,H]^T(bf16) + fcb
// 64 tiles of 16x16 -> 8 WGs x 8 waves
// ---------------------------------------------------------------------------
__global__ void __launch_bounds__(256) fc_kernel(
    const bf16* __restrict__ h, const bf16* __restrict__ W,
    const float* __restrict__ bias, float* __restrict__ out) {
  const int id = blockIdx.x * 8 + (threadIdx.x >> 5);   // 0..63
  const int lane = threadIdx.x & 31;
  const int m_base = (id & 3) * 16;
  const int n_base = (id >> 2) * 16;
  v8f c = {};
#pragma unroll 4
  for (int k = 0; k < RNN_H; k += 32) {
    v16bf a = load_A16x32(h, RNN_H, m_base, k, lane);
    v16bf b = load_B32x16(W, RNN_H, n_base, k, lane);
    c = wmma_bf16(a, b, c);
  }
  const int col = n_base + (lane & 15);
  const int r0 = m_base + (lane >> 4) * 8;
  float bv = bias[col];
#pragma unroll
  for (int v = 0; v < 8; ++v) out[(r0 + v) * RNN_O + col] = c[v] + bv;
}

// ---------------------------------------------------------------------------
// Launch
// ---------------------------------------------------------------------------
extern "C" void kernel_launch(void* const* d_in, const int* in_sizes, int n_in,
                              void* d_out, int out_size, void* d_ws, size_t ws_size,
                              hipStream_t stream) {
  (void)in_sizes; (void)n_in; (void)out_size; (void)ws_size;
  const float* x    = (const float*)d_in[0];
  const float* Wih0 = (const float*)d_in[1];
  const float* bih0 = (const float*)d_in[2];
  const float* Whh0 = (const float*)d_in[3];
  const float* bhh0 = (const float*)d_in[4];
  const float* Wih1 = (const float*)d_in[5];
  const float* bih1 = (const float*)d_in[6];
  const float* Whh1 = (const float*)d_in[7];
  const float* bhh1 = (const float*)d_in[8];
  const float* fcW  = (const float*)d_in[9];
  const float* fcb  = (const float*)d_in[10];
  float* out = (float*)d_out;

  constexpr int B = RNN_B, T = RNN_T, I = RNN_I, H = RNN_H, O = RNN_O;
  constexpr int M = B * T;

  char* ws = (char*)d_ws;
  size_t off = 0;
  auto wsalloc = [&](size_t bytes) -> char* {
    char* p = ws + off;
    off += (bytes + 255) & ~(size_t)255;
    return p;
  };
  unsigned* counter = (unsigned*)wsalloc(256);
  float* xp   = (float*)wsalloc((size_t)M * H * sizeof(float));
  bf16* hseq  = (bf16*)wsalloc((size_t)M * H * sizeof(bf16));
  bf16* xbf   = (bf16*)wsalloc((size_t)M * I * sizeof(bf16));
  bf16* Wih0b = (bf16*)wsalloc((size_t)H * I * sizeof(bf16));
  bf16* Whh0b = (bf16*)wsalloc((size_t)H * H * sizeof(bf16));
  bf16* Wih1b = (bf16*)wsalloc((size_t)H * H * sizeof(bf16));
  bf16* Whh1b = (bf16*)wsalloc((size_t)H * H * sizeof(bf16));
  bf16* fcWb  = (bf16*)wsalloc((size_t)O * H * sizeof(bf16));
  bf16* hp0   = (bf16*)wsalloc((size_t)B * H * sizeof(bf16));
  bf16* hp1   = (bf16*)wsalloc((size_t)B * H * sizeof(bf16));

  auto cvt = [&](const float* s, bf16* d, int n) {
    f32_to_bf16_kernel<<<(n / 4 + 255) / 256, 256, 0, stream>>>(s, d, n);
  };
  cvt(x,    xbf,   M * I);
  cvt(Wih0, Wih0b, H * I);
  cvt(Whh0, Whh0b, H * H);
  cvt(Whh1, Whh1b, H * H);
  cvt(Wih1, Wih1b, H * H);
  cvt(fcW,  fcWb,  O * H);

  // ---- Layer 0 ----
  gemm_bias_kernel<I><<<dim3(H / 64, M / 128), 256, 0, stream>>>(
      xbf, Wih0b, bih0, bhh0, xp, M, H);
  init_kernel<<<(B * H + 255) / 256, 256, 0, stream>>>(hp0, counter, B * H);
  rnn_persistent_kernel<<<RNN_NWG, 128, 0, stream>>>(
      xp, Whh0b, hp0, hp1, hseq, counter, T);

  // ---- Layer 1 ----
  gemm_bias_kernel<H><<<dim3(H / 64, M / 128), 256, 0, stream>>>(
      hseq, Wih1b, bih1, bhh1, xp, M, H);
  init_kernel<<<(B * H + 255) / 256, 256, 0, stream>>>(hp0, counter, B * H);
  rnn_persistent_kernel<<<RNN_NWG, 128, 0, stream>>>(
      xp, Whh1b, hp0, hp1, hseq, counter, T);

  // ---- FC head: T=512 is even, so final h lives in hp0 ----
  fc_kernel<<<8, 256, 0, stream>>>(hp0, fcWb, fcb, out);
}
